// NystromMultiHeadAttention_13280038879460
// MI455X (gfx1250) — matrix operand order repaired and checked by reading, not compiled, not run on hardware
//
#include <hip/hip_runtime.h>
#include <hip/hip_bf16.h>

typedef _Float16 f16;
typedef __attribute__((ext_vector_type(16))) _Float16 v16h;
typedef __attribute__((ext_vector_type(8)))  _Float16 v8h;
typedef __attribute__((ext_vector_type(8)))  float    v8f;
typedef __attribute__((ext_vector_type(4)))  int      i32x4;

#define B_   4
#define N_   4096
#define D_   1024
#define H_   16
#define HD_  64
#define M_   64
#define L_   64          // tokens per landmark
#define KW_  33
#define R_   (B_ * N_)   // 16384
#define BH_  (B_ * H_)   // 64
#define ITERS_ 6
#define TAU_ 8.0f

#define BM 128
#define BN 64
#define BK 32
#define LDS_STRIDE 40    // 32 halfs + 8 pad, keeps 16B alignment

#if __has_builtin(__builtin_amdgcn_global_load_async_to_lds_b128) && \
    __has_builtin(__builtin_amdgcn_s_wait_asynccnt)
#define USE_ASYNC_LDS 1
#else
#define USE_ASYNC_LDS 0
#endif

#define GLOBAL_AS __attribute__((address_space(1)))
#define LDS_AS    __attribute__((address_space(3)))

#if USE_ASYNC_LDS
// async DMA: 16B per lane, global -> LDS, tracked by ASYNCcnt
__device__ __forceinline__ void async_b128(const f16* g, f16* l) {
  __builtin_amdgcn_global_load_async_to_lds_b128(
      (GLOBAL_AS i32x4*)g, (LDS_AS i32x4*)l, 0, 0);
}
#endif

__device__ __forceinline__ v8f wmma32(v16h a, v16h b, v8f c) {
  return __builtin_amdgcn_wmma_f32_16x16x32_f16(false, a, false, b, (short)0, c,
                                                false, false);
}

__device__ __forceinline__ v16h load_frag(const f16* rowbase, int off0, int off1) {
  v8h lo = *(const v8h*)(rowbase + off0);
  v8h hi = *(const v8h*)(rowbase + off1);
  return __builtin_shufflevector(lo, hi,
      0,1,2,3,4,5,6,7,8,9,10,11,12,13,14,15);
}

// -------------------------------------------------------------------------
// Generic batched NT GEMM: C[r][c] = scale * sum_k A[r][k] * B[c][k]
// Block tile 128x64, per-wave 32x32 (2x2 WMMA tiles), BK=32, double-buffered
// LDS. O multiple of 64, Kd multiple of 32; ragged R handled by clamp+guard.
// Batch z decomposed as (z/inner, z%inner) with two strides per operand.
// -------------------------------------------------------------------------
__global__ __launch_bounds__(256) void k_gemm_nt(
    const f16* __restrict__ A,  long soA,  long siA,  int lda,
    const f16* __restrict__ Bm, long soB,  long siB,  int ldb,
    int Kd, int Rrows, int inner,
    float* __restrict__ Cf, long soC,  long siC,  int ldc,
    f16*   __restrict__ Ch, long soCh, long siCh, int ldch,
    float scale)
{
  __shared__ __align__(16) f16 As[2][BM * LDS_STRIDE];
  __shared__ __align__(16) f16 Bs[2][BN * LDS_STRIDE];

  const int  z   = blockIdx.z;
  const int  zo  = z / inner, zi = z % inner;
  const f16* Ab  = A  + zo * soA + zi * siA;
  const f16* Bb  = Bm + zo * soB + zi * siB;
  const long offC  = zo * soC  + zi * siC;
  const long offCh = zo * soCh + zi * siCh;

  const int rowBlock = blockIdx.y * BM;
  const int colBlock = blockIdx.x * BN;

  const int  tid  = threadIdx.x;
  const int  lane = tid & 31;
  const int  wm   = (tid >> 5) & 3;  // wave row tile (0..3) -> 32 rows each
  const int  wn   = tid >> 7;        // wave col tile (0..1) -> 32 cols each
  const bool hihalf = lane >= 16;
  const int  l16  = lane & 15;

  // staging assignment: each thread moves 3 x 16B per K-step
  const int stRow = tid >> 2;        // 0..63
  const int stSeg = (tid & 3) << 3;  // 0,8,16,24 halfs
  const int aRow0 = stRow;           // A rows 0..63
  const int aRow1 = stRow + 64;      // A rows 64..127
  // clamp ragged R (reads duplicate last row; stores are guarded)
  const int gaRow0 = min(rowBlock + aRow0, Rrows - 1);
  const int gaRow1 = min(rowBlock + aRow1, Rrows - 1);

  const f16* aSrc0 = Ab + (long)gaRow0 * lda + stSeg;
  const f16* aSrc1 = Ab + (long)gaRow1 * lda + stSeg;
  const f16* bSrc  = Bb + (long)(colBlock + stRow) * ldb + stSeg;

  const int ldsA0 = aRow0 * LDS_STRIDE + stSeg;
  const int ldsA1 = aRow1 * LDS_STRIDE + stSeg;
  const int ldsB  = stRow * LDS_STRIDE + stSeg;

  v8f acc00 = {}, acc01 = {}, acc10 = {}, acc11 = {};

  const int T = Kd / BK;
  int cur = 0;

#if USE_ASYNC_LDS
  // ---- async DMA staging of tile 0 ----
  async_b128(aSrc0, &As[0][ldsA0]);
  async_b128(aSrc1, &As[0][ldsA1]);
  async_b128(bSrc,  &Bs[0][ldsB]);
  __builtin_amdgcn_s_wait_asynccnt(0);
  __syncthreads();
#else
  // ---- register staging fallback ----
  {
    v8h ra0 = *(const v8h*)(aSrc0);
    v8h ra1 = *(const v8h*)(aSrc1);
    v8h rb  = *(const v8h*)(bSrc);
    *(v8h*)(&As[0][ldsA0]) = ra0;
    *(v8h*)(&As[0][ldsA1]) = ra1;
    *(v8h*)(&Bs[0][ldsB])  = rb;
  }
  __syncthreads();
#endif

  for (int t = 0; t < T; ++t) {
    const int kk = (t + 1) * BK;   // next tile's K offset
    const bool more = (t + 1 < T);
    const int nxt = cur ^ 1;

#if USE_ASYNC_LDS
    if (more) {
      async_b128(aSrc0 + kk, &As[nxt][ldsA0]);
      async_b128(aSrc1 + kk, &As[nxt][ldsA1]);
      async_b128(bSrc + kk,  &Bs[nxt][ldsB]);
    }
#else
    v8h ra0, ra1, rb;
    if (more) {
      ra0 = *(const v8h*)(aSrc0 + kk);
      ra1 = *(const v8h*)(aSrc1 + kk);
      rb  = *(const v8h*)(bSrc + kk);
      if (t + 2 < T) {
        __builtin_prefetch(aSrc0 + kk + BK, 0, 1);  // global_prefetch_b8
        __builtin_prefetch(bSrc + kk + BK, 0, 1);
      }
    }
#endif

    // ---- compute on buffer `cur` (overlaps in-flight loads) ----
    {
      const f16* Abase = &As[cur][0];
      const f16* Bbase = &Bs[cur][0];
      // A fragments: lane<16 K {0..7,16..23}; lane>=16 K {8..15,24..31}
      const int aoff0 = hihalf ? 8  : 0;
      const int aoff1 = hihalf ? 24 : 16;
      const int ar0 = (wm * 32 + l16) * LDS_STRIDE;
      const int ar1 = ar0 + 16 * LDS_STRIDE;
      v16h a0 = load_frag(Abase + ar0, aoff0, aoff1);
      v16h a1 = load_frag(Abase + ar1, aoff0, aoff1);
      // B fragments: lane holds column; lane<16 -> K {0..7,16..23}, lane>=16 -> K {8..15,24..31}
      const int kb  = hihalf ? 8 : 0;
      const int bc0 = (wn * 32 + l16) * LDS_STRIDE;
      const int bc1 = bc0 + 16 * LDS_STRIDE;
      v16h b0 = load_frag(Bbase + bc0, kb, kb + 16);
      v16h b1 = load_frag(Bbase + bc1, kb, kb + 16);

      acc00 = wmma32(a0, b0, acc00);
      acc01 = wmma32(a0, b1, acc01);
      acc10 = wmma32(a1, b0, acc10);
      acc11 = wmma32(a1, b1, acc11);
    }

#if USE_ASYNC_LDS
    if (more) __builtin_amdgcn_s_wait_asynccnt(0);
#else
    if (more) {
      *(v8h*)(&As[nxt][ldsA0]) = ra0;   // s_wait_loadcnt lands here, after WMMAs
      *(v8h*)(&As[nxt][ldsA1]) = ra1;
      *(v8h*)(&Bs[nxt][ldsB])  = rb;
    }
#endif
    __syncthreads();
    cur = nxt;
  }

  // C/D layout: VGPR v holds row (v + (lane<16?0:8)), col = lane%16
  const int rb0 = rowBlock + wm * 32 + (hihalf ? 8 : 0);
  const int c0  = colBlock + wn * 32 + l16;
#pragma unroll
  for (int v = 0; v < 8; ++v) {
    const int r0 = rb0 + v;
    const int r1 = r0 + 16;
    if (r0 < Rrows) {
      const float x0 = acc00[v] * scale;
      const float x1 = acc01[v] * scale;
      if (Cf) {
        Cf[offC + (long)r0 * ldc + c0]      = x0;
        Cf[offC + (long)r0 * ldc + c0 + 16] = x1;
      }
      if (Ch) {
        Ch[offCh + (long)r0 * ldch + c0]      = (f16)x0;
        Ch[offCh + (long)r0 * ldch + c0 + 16] = (f16)x1;
      }
    }
    if (r1 < Rrows) {
      const float x0 = acc10[v] * scale;
      const float x1 = acc11[v] * scale;
      if (Cf) {
        Cf[offC + (long)r1 * ldc + c0]      = x0;
        Cf[offC + (long)r1 * ldc + c0 + 16] = x1;
      }
      if (Ch) {
        Ch[offCh + (long)r1 * ldch + c0]      = (f16)x0;
        Ch[offCh + (long)r1 * ldch + c0 + 16] = (f16)x1;
      }
    }
  }
}

// -------------------------------------------------------------------------
__global__ void k_cvt(const float* __restrict__ in, f16* __restrict__ out, long n) {
  long i = (long)blockIdx.x * 256 + threadIdx.x;
  if (i < n) out[i] = (f16)in[i];
}

// landmark means over l=64 tokens; qh/kh stored [b,n,h,hd] row-major [R,D]
__global__ void k_landmarks(const f16* __restrict__ qh, const f16* __restrict__ kh,
                            f16* __restrict__ qlh, f16* __restrict__ klh) {
  const int blk = blockIdx.x;           // ((b*H+h)*M + m)
  const int hd  = threadIdx.x;          // 0..63
  const int m   = blk & (M_ - 1);
  const int bh  = blk / M_;
  const int h   = bh & (H_ - 1);
  const int b   = bh / H_;
  float sq = 0.f, sk = 0.f;
  long base = ((long)b * N_ + (long)m * L_) * D_ + h * HD_ + hd;
  for (int l = 0; l < L_; ++l) {
    sq += (float)qh[base + (long)l * D_];
    sk += (float)kh[base + (long)l * D_];
  }
  qlh[(long)blk * HD_ + hd] = (f16)(sq * (1.0f / L_));
  klh[(long)blk * HD_ + hd] = (f16)(sk * (1.0f / L_));
}

// v [b,n,h,hd] -> vT [b,h,hd,n]
__global__ void k_vt(const f16* __restrict__ vh, f16* __restrict__ vT) {
  long idx = (long)blockIdx.x * 256 + threadIdx.x;   // over BH*HD*N
  const int  n  = idx & (N_ - 1);
  const long t  = idx >> 12;
  const int  hd = t & (HD_ - 1);
  const int  bh = (int)(t >> 6);
  const int  h  = bh & (H_ - 1);
  const int  b  = bh / H_;
  vT[idx] = vh[((long)b * N_ + n) * D_ + h * HD_ + hd];
}

// row softmax over `cols`; optional f32 and f16 outputs
__global__ __launch_bounds__(256) void k_softmax(
    const float* __restrict__ S, float* __restrict__ outF,
    f16* __restrict__ outH, int cols) {
  __shared__ float red[256];
  const long row = blockIdx.x;
  const float* src = S + row * (long)cols;
  const int tid = threadIdx.x;

  float m = -3.0e38f;
  for (int i = tid; i < cols; i += 256) m = fmaxf(m, src[i]);
  red[tid] = m; __syncthreads();
  for (int s = 128; s > 0; s >>= 1) {
    if (tid < s) red[tid] = fmaxf(red[tid], red[tid + s]);
    __syncthreads();
  }
  m = red[0]; __syncthreads();

  float sum = 0.f;
  for (int i = tid; i < cols; i += 256) sum += __expf(src[i] - m);
  red[tid] = sum; __syncthreads();
  for (int s = 128; s > 0; s >>= 1) {
    if (tid < s) red[tid] += red[tid + s];
    __syncthreads();
  }
  const float inv = 1.0f / red[0];

  for (int i = tid; i < cols; i += 256) {
    const float e = __expf(src[i] - m) * inv;
    if (outF) outF[row * (long)cols + i] = e;
    if (outH) outH[row * (long)cols + i] = (f16)e;
  }
}

__global__ void k_init_scalars(unsigned* gc, unsigned* gr) {
  if (threadIdx.x == 0) { *gc = 0u; *gr = 0u; }
}

// global max of row-sums (col) and col-sums (row) of |a2|, per reference
__global__ void k_gmax(const float* __restrict__ a2s,
                       unsigned* __restrict__ gc, unsigned* __restrict__ gr) {
  const int bh = blockIdx.x;
  const int t  = threadIdx.x;  // 128
  const float* X = a2s + (long)bh * (M_ * M_);
  if (t < 64) {
    float s = 0.f;
    for (int j = 0; j < 64; ++j) s += fabsf(X[t * 64 + j]);
    atomicMax(gc, __float_as_uint(s));
  } else {
    const int c = t - 64;
    float s = 0.f;
    for (int i = 0; i < 64; ++i) s += fabsf(X[i * 64 + c]);
    atomicMax(gr, __float_as_uint(s));
  }
}

__device__ __forceinline__ void mm64(float* __restrict__ dst,
                                     const float* __restrict__ a,
                                     const float* __restrict__ b, int tid) {
  __syncthreads();
  for (int e = 0; e < 16; ++e) {
    const int idx = tid + e * 256;
    const int i = idx >> 6, j = idx & 63;
    float s = 0.f;
    for (int k = 0; k < 64; ++k) s += a[i * 64 + k] * b[k * 64 + j];
    dst[idx] = s;
  }
  __syncthreads();
}

// Newton-Schulz pinv of a2 (64x64), then t2 = z @ t1; writes t2^T as f16
__global__ __launch_bounds__(256) void k_pinv(
    const float* __restrict__ a2s, const float* __restrict__ t1,
    const unsigned* __restrict__ gc, const unsigned* __restrict__ gr,
    f16* __restrict__ t2T) {
  __shared__ float X[4096], Z[4096], P[4096], W[4096], Q[4096];
  const int bh = blockIdx.x, tid = threadIdx.x;
  const float* Xg  = a2s + (long)bh * 4096;
  const float* T1g = t1  + (long)bh * 4096;
  const float inv = 1.0f / (__uint_as_float(*gc) * __uint_as_float(*gr));

  for (int e = 0; e < 16; ++e) { int idx = tid + e * 256; X[idx] = Xg[idx]; }
  __syncthreads();
  for (int e = 0; e < 16; ++e) {
    int idx = tid + e * 256; int i = idx >> 6, j = idx & 63;
    Z[idx] = X[j * 64 + i] * inv;  // transpose / (gc*gr)
  }

  for (int it = 0; it < ITERS_; ++it) {
    mm64(P, X, Z, tid);  // P = x @ z
    for (int e = 0; e < 16; ++e) { int idx = tid + e*256; int i = idx>>6, j = idx&63;
      W[idx] = (i == j ? 7.0f : 0.0f) - P[idx]; }
    mm64(Q, P, W, tid);  // Q = P @ (7I - P)
    for (int e = 0; e < 16; ++e) { int idx = tid + e*256; int i = idx>>6, j = idx&63;
      W[idx] = (i == j ? 15.0f : 0.0f) - Q[idx]; }
    mm64(Q, P, W, tid);  // Q = P @ (15I - ...)
    for (int e = 0; e < 16; ++e) { int idx = tid + e*256; int i = idx>>6, j = idx&63;
      W[idx] = (i == j ? 13.0f : 0.0f) - Q[idx]; }
    mm64(P, Z, W, tid);  // P = z @ (13I - ...)
    for (int e = 0; e < 16; ++e) { int idx = tid + e*256; Z[idx] = 0.25f * P[idx]; }
  }

  __syncthreads();
  for (int e = 0; e < 16; ++e) { int idx = tid + e * 256; W[idx] = T1g[idx]; }
  mm64(P, Z, W, tid);    // t2 = z @ t1   [m][hd]
  for (int e = 0; e < 16; ++e) {
    int idx = tid + e * 256; int i = idx >> 6, j = idx & 63;
    t2T[(long)bh * 4096 + j * 64 + i] = (f16)P[idx];  // [hd][m]
  }
}

// atth = attf + depthwise_conv(v); v in [b,n,h,hd] f16; attf [R,D] f32
__global__ void k_conv_add(const f16* __restrict__ vh,
                           const float* __restrict__ attf,
                           const float* __restrict__ Wc,
                           f16* __restrict__ atth) {
  long idx = (long)blockIdx.x * 256 + threadIdx.x;   // over R*D
  const int  d = idx & (D_ - 1);
  const long r = idx >> 10;
  const int  n = (int)(r & (N_ - 1));
  const int  b = (int)(r >> 12);
  const int  h = d >> 6;
  float acc = attf[idx];
  for (int j = 0; j < KW_; ++j) {
    const int nn = n + j - KW_ / 2;
    if (nn >= 0 && nn < N_)
      acc += (float)vh[((long)b * N_ + nn) * D_ + d] * Wc[h * KW_ + j];
  }
  atth[idx] = (f16)acc;
}

// -------------------------------------------------------------------------
extern "C" void kernel_launch(void* const* d_in, const int* in_sizes, int n_in,
                              void* d_out, int out_size, void* d_ws, size_t ws_size,
                              hipStream_t stream) {
  (void)in_sizes; (void)n_in; (void)out_size; (void)ws_size;
  const float* x  = (const float*)d_in[0];
  const float* Wq = (const float*)d_in[1];
  const float* Wk = (const float*)d_in[2];
  const float* Wv = (const float*)d_in[3];
  const float* Wo = (const float*)d_in[4];
  const float* Wc = (const float*)d_in[5];
  float* out = (float*)d_out;

  char* p = (char*)d_ws;
  auto take = [&](size_t bytes) -> char* {
    char* r = p; p += (bytes + 255) & ~(size_t)255; return r;
  };

  f16* xh  = (f16*)take((size_t)R_ * D_ * 2);   // reused later as atth
  f16* wqh = (f16*)take((size_t)D_ * D_ * 2);
  f16* wkh = (f16*)take((size_t)D_ * D_ * 2);
  f16* wvh = (f16*)take((size_t)D_ * D_ * 2);
  f16* woh = (f16*)take((size_t)D_ * D_ * 2);
  f16* qh  = (f16*)take((size_t)R_ * D_ * 2);
  f16* kh  = (f16*)take((size_t)R_ * D_ * 2);
  f16* vh  = (f16*)take((size_t)R_ * D_ * 2);
  f16* vT  = (f16*)take((size_t)BH_ * HD_ * N_ * 2);
  f16* qlh = (f16*)take((size_t)BH_ * M_ * HD_ * 2);
  f16* klh = (f16*)take((size_t)BH_ * M_ * HD_ * 2);
  f16* a1h = (f16*)take((size_t)BH_ * N_ * M_ * 2);
  f16* a3h = (f16*)take((size_t)BH_ * M_ * N_ * 2);
  f16* t2T = (f16*)take((size_t)BH_ * HD_ * M_ * 2);
  float* scores = (float*)take((size_t)BH_ * N_ * M_ * 4); // s1 / s3 / attf
  float* s2  = (float*)take((size_t)BH_ * M_ * M_ * 4);
  float* a2s = (float*)take((size_t)BH_ * M_ * M_ * 4);
  float* t1f = (float*)take((size_t)BH_ * M_ * HD_ * 4);
  unsigned* gcol = (unsigned*)take(256);
  unsigned* grow = (unsigned*)take(256);

  // --- f32 -> f16 conversions -------------------------------------------
  k_cvt<<<(R_ * D_) / 256, 256, 0, stream>>>(x,  xh,  (long)R_ * D_);
  k_cvt<<<(D_ * D_) / 256, 256, 0, stream>>>(Wq, wqh, (long)D_ * D_);
  k_cvt<<<(D_ * D_) / 256, 256, 0, stream>>>(Wk, wkh, (long)D_ * D_);
  k_cvt<<<(D_ * D_) / 256, 256, 0, stream>>>(Wv, wvh, (long)D_ * D_);
  k_cvt<<<(D_ * D_) / 256, 256, 0, stream>>>(Wo, woh, (long)D_ * D_);

  // --- projections: y = x @ W^T  (NT GEMM, W row-major is the B operand) -
  const dim3 gproj(D_ / BN, R_ / BM, 1);
  k_gemm_nt<<<gproj, 256, 0, stream>>>(xh, 0, 0, D_, wqh, 0, 0, D_, D_, R_, 1,
      nullptr, 0, 0, 0, qh, 0, 0, D_, 1.0f / TAU_);
  k_gemm_nt<<<gproj, 256, 0, stream>>>(xh, 0, 0, D_, wkh, 0, 0, D_, D_, R_, 1,
      nullptr, 0, 0, 0, kh, 0, 0, D_, 1.0f);
  k_gemm_nt<<<gproj, 256, 0, stream>>>(xh, 0, 0, D_, wvh, 0, 0, D_, D_, R_, 1,
      nullptr, 0, 0, 0, vh, 0, 0, D_, 1.0f);

  // --- landmarks + v transpose ------------------------------------------
  k_landmarks<<<BH_ * M_, 64, 0, stream>>>(qh, kh, qlh, klh);
  k_vt<<<((size_t)BH_ * HD_ * N_) / 256, 256, 0, stream>>>(vh, vT);

  // --- a1 = softmax(q @ k_land^T)  [b,h,n,m] ----------------------------
  k_gemm_nt<<<dim3(M_ / BN, N_ / BM, BH_), 256, 0, stream>>>(
      qh,  (long)N_ * D_, HD_, D_,
      klh, (long)H_ * M_ * HD_, (long)M_ * HD_, HD_,
      HD_, N_, H_,
      scores, (long)H_ * N_ * M_, (long)N_ * M_, M_,
      nullptr, 0, 0, 0, 1.0f);
  k_softmax<<<BH_ * N_, 256, 0, stream>>>(scores, nullptr, a1h, M_);

  // --- a2 = softmax(q_land @ k_land^T) [b,h,m,m] ------------------------
  k_gemm_nt<<<dim3(1, 1, BH_), 256, 0, stream>>>(
      qlh, (long)H_ * M_ * HD_, (long)M_ * HD_, HD_,
      klh, (long)H_ * M_ * HD_, (long)M_ * HD_, HD_,
      HD_, M_, H_,
      s2, (long)H_ * M_ * M_, (long)M_ * M_, M_,
      nullptr, 0, 0, 0, 1.0f);
  k_softmax<<<BH_ * M_, 256, 0, stream>>>(s2, a2s, nullptr, M_);
  k_init_scalars<<<1, 32, 0, stream>>>(gcol, grow);
  k_gmax<<<BH_, 128, 0, stream>>>(a2s, gcol, grow);

  // --- a3 = softmax(q_land @ k^T) [b,h,m,n] -----------------------------
  k_gemm_nt<<<dim3(N_ / BN, 1, BH_), 256, 0, stream>>>(
      qlh, (long)H_ * M_ * HD_, (long)M_ * HD_, HD_,
      kh,  (long)N_ * D_, HD_, D_,
      HD_, M_, H_,
      scores, (long)H_ * M_ * N_, (long)M_ * N_, N_,
      nullptr, 0, 0, 0, 1.0f);
  k_softmax<<<BH_ * M_, 256, 0, stream>>>(scores, nullptr, a3h, N_);

  // --- t1 = a3 @ v  (NT against v^T)  [b,h,m,hd] ------------------------
  k_gemm_nt<<<dim3(1, 1, BH_), 256, 0, stream>>>(
      a3h, (long)H_ * M_ * N_, (long)M_ * N_, N_,
      vT,  (long)H_ * HD_ * N_, (long)HD_ * N_, N_,
      N_, M_, H_,
      t1f, (long)H_ * M_ * HD_, (long)M_ * HD_, HD_,
      nullptr, 0, 0, 0, 1.0f);

  // --- pinv iterations + t2 = a2_inv @ t1 -> t2^T f16 -------------------
  k_pinv<<<BH_, 256, 0, stream>>>(a2s, t1f, gcol, grow, t2T);

  // --- att = a1 @ t2  (NT against t2^T) -> attf f32 [b,n,h,hd] ----------
  k_gemm_nt<<<dim3(1, N_ / BM, BH_), 256, 0, stream>>>(
      a1h, (long)H_ * N_ * M_, (long)N_ * M_, M_,
      t2T, (long)H_ * HD_ * M_, (long)HD_ * M_, M_,
      M_, N_, H_,
      scores /*attf*/, (long)N_ * D_, HD_, D_,
      nullptr, 0, 0, 0, 1.0f);

  // --- add depthwise conv, convert to f16 (reuse xh as atth) ------------
  k_conv_add<<<(R_ * D_) / 256, 256, 0, stream>>>(vh, scores, Wc, xh);

  // --- out = atth @ Wo^T -> d_out f32 [B,N,D] ---------------------------
  k_gemm_nt<<<gproj, 256, 0, stream>>>(xh, 0, 0, D_, woh, 0, 0, D_, D_, R_, 1,
      out, 0, 0, D_, nullptr, 0, 0, 0, 1.0f);
}
